// MoELayer_11544872092304
// MI455X (gfx1250) — matrix operand order repaired and checked by reading, not compiled, not run on hardware
//
#include <hip/hip_runtime.h>

// ---------------------------------------------------------------------------
// MoE layer for MI455X (gfx1250, wave32, WMMA).
// Routed top-2 compute (137 GFLOP instead of dense 550 GFLOP), bf16 WMMA with
// f32 accumulation. Weights converted once to bf16-transposed (67MB, L2
// resident) so B-fragments are contiguous 32B loads. Per-(token,k) contrib
// buffer keeps the hot path atomic-free and deterministic.
// ---------------------------------------------------------------------------

typedef __bf16 bf16_t;
typedef __attribute__((ext_vector_type(16))) __bf16 v16bf;
typedef __attribute__((ext_vector_type(8)))  __bf16 v8bf;
typedef __attribute__((ext_vector_type(4)))  __bf16 v4bf;
typedef __attribute__((ext_vector_type(8)))  float  v8f;
typedef __attribute__((ext_vector_type(4)))  float  v4f;

constexpr int NT   = 8192;   // tokens (B*S)
constexpr int DM   = 1024;   // model dim
constexpr int EXP  = 8;      // experts
constexpr int HID  = 2048;   // hidden dim
constexpr int TM   = 16;     // token tile (WMMA M)
constexpr int HC   = 128;    // hidden chunk per iteration
constexpr int TILES_PER_E = NT / TM;  // 512

// ---------------- workspace layout (bytes, 256-aligned) ----------------
constexpr size_t WS_XBF   = 0;                                   // NT*DM bf16
constexpr size_t WS_W1T   = WS_XBF  + (size_t)NT * DM * 2;       // [E][H][D] bf16
constexpr size_t WS_W2T   = WS_W1T  + (size_t)EXP * HID * DM * 2;// [E][D][H] bf16
constexpr size_t WS_CONTR = WS_W2T  + (size_t)EXP * DM * HID * 2;// [NT][2][DM] f32
constexpr size_t WS_WGT   = WS_CONTR + (size_t)NT * 2 * DM * 4;  // [NT][2] f32
constexpr size_t WS_LIST  = WS_WGT  + (size_t)NT * 2 * 4;        // [E][NT] int
constexpr size_t WS_CNT   = WS_LIST + (size_t)EXP * NT * 4;      // [E] int
constexpr size_t WS_PSUM  = WS_CNT  + 256;                       // [E] f32

// ---------------------------------------------------------------------------
__global__ void zero_gate_state(int* cnt, float* psum) {
    int t = threadIdx.x;
    if (t < EXP) { cnt[t] = 0; psum[t] = 0.f; }
}

// x fp32 -> bf16
__global__ void cvt_x_kernel(const float* __restrict__ x, bf16_t* __restrict__ xb) {
    int i = blockIdx.x * 256 + threadIdx.x;           // over NT*DM/4
    v4f v = *(const v4f*)(x + (size_t)i * 4);
    v4bf o = { (bf16_t)v.x, (bf16_t)v.y, (bf16_t)v.z, (bf16_t)v.w };
    *(v4bf*)(xb + (size_t)i * 4) = o;
}

// src: [Z][R][C] f32  ->  dst: [Z][C][R] bf16  (tiled LDS transpose)
__global__ void transpose_cvt_kernel(const float* __restrict__ src,
                                     bf16_t* __restrict__ dst, int R, int C) {
    __shared__ float tile[32][33];
    const int z  = blockIdx.z;
    const int c0 = blockIdx.x * 32, r0 = blockIdx.y * 32;
    const int tx = threadIdx.x & 31, ty = threadIdx.x >> 5;  // 32 x 8
    const float* s = src + (size_t)z * R * C;
    bf16_t*      d = dst + (size_t)z * R * C;
#pragma unroll
    for (int i = 0; i < 4; i++) {
        int r = ty + i * 8;
        tile[r][tx] = s[(size_t)(r0 + r) * C + c0 + tx];
    }
    __syncthreads();
#pragma unroll
    for (int i = 0; i < 4; i++) {
        int c = ty + i * 8;
        d[(size_t)(c0 + c) * R + r0 + tx] = (bf16_t)tile[tx][c];
    }
}

// ---------------- gate: one wave per token ----------------
__global__ __launch_bounds__(256)
void gate_kernel(const float* __restrict__ x, const float* __restrict__ gate_w,
                 const float* __restrict__ gate_b, int* __restrict__ tok_list,
                 float* __restrict__ wgt, int* __restrict__ cnt,
                 float* __restrict__ psum) {
    const int n    = blockIdx.x * 8 + (threadIdx.x >> 5);
    const int lane = threadIdx.x & 31;
    float p[EXP];
#pragma unroll
    for (int e = 0; e < EXP; e++) p[e] = 0.f;
    const float* xr = x + (size_t)n * DM;
    for (int d = lane; d < DM; d += 32) {
        float xv = xr[d];
        const float* g = gate_w + (size_t)d * EXP;
#pragma unroll
        for (int e = 0; e < EXP; e++) p[e] = fmaf(xv, g[e], p[e]);
    }
#pragma unroll
    for (int e = 0; e < EXP; e++)
        for (int off = 16; off; off >>= 1) p[e] += __shfl_xor(p[e], off, 32);

    if (lane == 0) {
        float lg[EXP], mx = -1e30f;
#pragma unroll
        for (int e = 0; e < EXP; e++) { lg[e] = p[e] + gate_b[e]; mx = fmaxf(mx, lg[e]); }
        float se = 0.f, gw8[EXP];
#pragma unroll
        for (int e = 0; e < EXP; e++) { gw8[e] = __expf(lg[e] - mx); se += gw8[e]; }
        float inv = 1.f / se;
#pragma unroll
        for (int e = 0; e < EXP; e++) { gw8[e] *= inv; atomicAdd(&psum[e], gw8[e]); }
        int i0 = 0;
#pragma unroll
        for (int e = 1; e < EXP; e++) if (gw8[e] > gw8[i0]) i0 = e;
        int i1 = (i0 == 0) ? 1 : 0;
#pragma unroll
        for (int e = 0; e < EXP; e++) if (e != i0 && gw8[e] > gw8[i1]) i1 = e;
        float s2 = gw8[i0] + gw8[i1];
        wgt[n * 2 + 0] = gw8[i0] / s2;
        wgt[n * 2 + 1] = gw8[i1] / s2;
        int p0 = atomicAdd(&cnt[i0], 1); tok_list[(size_t)i0 * NT + p0] = n * 2 + 0;
        int p1 = atomicAdd(&cnt[i1], 1); tok_list[(size_t)i1 * NT + p1] = n * 2 + 1;
    }
}

// ---------------- expert FFN: 16-token tile, 8 waves, bf16 WMMA ----------------
__device__ inline v16bf make_a(v8bf lo, v8bf hi) {
    return __builtin_shufflevector(lo, hi, 0,1,2,3,4,5,6,7,8,9,10,11,12,13,14,15);
}

__global__ __launch_bounds__(256)
void expert_ffn_kernel(const bf16_t* __restrict__ x_bf, const bf16_t* __restrict__ w1t,
                       const bf16_t* __restrict__ w2t, const float* __restrict__ b1,
                       const float* __restrict__ b2, const int* __restrict__ tok_list,
                       const int* __restrict__ cnt, float* __restrict__ contrib) {
    const int e    = blockIdx.x >> 9;          // 512 tiles / expert
    const int tile = blockIdx.x & (TILES_PER_E - 1);
    const int count = cnt[e];
    if (tile * TM >= count) return;

    __shared__ __align__(16) bf16_t xs[TM][DM + 8];  // 16 x 1032 bf16 (~33 KB)
    __shared__ __align__(16) bf16_t hs[TM][HC + 8];  // 16 x 136 bf16
    __shared__ int s_slot[TM];

    const int tid  = threadIdx.x;
    const int wave = tid >> 5;
    const int lane = tid & 31;

    if (tid < TM) {
        int r = tile * TM + tid;
        s_slot[tid] = (r < count) ? tok_list[(size_t)e * NT + r] : -1;
    }
    __syncthreads();

    // gather 16 token rows into LDS as bf16 (16B chunks)
    for (int c = tid; c < TM * (DM / 8); c += 256) {
        int row = c / (DM / 8);
        int off = (c % (DM / 8)) * 8;
        int entry = s_slot[row];
        int n = (entry >= 0) ? (entry >> 1) : 0;
        *(v8bf*)(&xs[row][off]) = *(const v8bf*)(x_bf + (size_t)n * DM + off);
    }

    const int mrow    = lane & 15;
    const int halfsel = (lane < 16) ? 0 : 16;   // A-fragment K base
    const int bkoff   = (lane < 16) ? 0 : 16;   // B-fragment K base
    const int bcol    = lane & 15;              // B/C column within 16

    const bf16_t* w1e = w1t + (size_t)e * HID * DM;  // [H][D]
    const bf16_t* w2e = w2t + (size_t)e * DM * HID;  // [D][H]

    const v8f vzero = {0.f,0.f,0.f,0.f,0.f,0.f,0.f,0.f};
    v8f acc[8];
#pragma unroll
    for (int t = 0; t < 8; t++) acc[t] = vzero;

    for (int hc = 0; hc < HID; hc += HC) {
        // ---- GEMM1: this wave computes h columns [hc+wave*16, +16), K=DM ----
        const int hcol0 = hc + wave * 16;
        v8f hacc = vzero;
        const bf16_t* bp1 = w1e + (size_t)(hcol0 + bcol) * DM + bkoff;
#pragma unroll 8
        for (int k0 = 0; k0 < DM; k0 += 32) {
            v8bf alo = *(const v8bf*)(&xs[mrow][k0 + halfsel]);
            v8bf ahi = *(const v8bf*)(&xs[mrow][k0 + halfsel + 8]);
            v16bf a = make_a(alo, ahi);
            v16bf b = *(const v16bf*)(bp1 + k0);
            hacc = __builtin_amdgcn_wmma_f32_16x16x32_bf16(
                false, a, false, b, (short)0, hacc, false, false);
        }
        // bias + relu + cvt bf16 -> LDS (C layout: VGPR v -> M, lane&15 -> N)
        __syncthreads();   // previous hs fully consumed
        float bias = b1[(size_t)e * HID + hcol0 + bcol];
#pragma unroll
        for (int v = 0; v < 8; v++) {
            int m = v + ((lane < 16) ? 0 : 8);
            float hv = hacc[v] + bias;
            hs[m][wave * 16 + bcol] = (bf16_t)(hv > 0.f ? hv : 0.f);
        }
        __syncthreads();   // hs ready for all waves

        // ---- GEMM2: acc[16 x 128 D-slice] += hs[16 x 128] @ w2 ----
        const int dcol0 = wave * 128;
#pragma unroll
        for (int k0 = 0; k0 < HC; k0 += 32) {
            v8bf alo = *(const v8bf*)(&hs[mrow][k0 + halfsel]);
            v8bf ahi = *(const v8bf*)(&hs[mrow][k0 + halfsel + 8]);
            v16bf a = make_a(alo, ahi);
#pragma unroll
            for (int t = 0; t < 8; t++) {
                const bf16_t* bp2 =
                    w2e + (size_t)(dcol0 + t * 16 + bcol) * HID + hc + k0 + bkoff;
                v16bf b = *(const v16bf*)bp2;
                acc[t] = __builtin_amdgcn_wmma_f32_16x16x32_bf16(
                    false, a, false, b, (short)0, acc[t], false, false);
            }
        }
    }

    // epilogue: contrib[(n*2+k)][col] = acc + b2  (atomic-free, deterministic)
#pragma unroll
    for (int t = 0; t < 8; t++) {
        int col = wave * 128 + t * 16 + bcol;
        float bb = b2[(size_t)e * DM + col];
#pragma unroll
        for (int v = 0; v < 8; v++) {
            int m = v + ((lane < 16) ? 0 : 8);
            int entry = s_slot[m];
            if (entry >= 0)
                contrib[(size_t)entry * DM + col] = acc[t][v] + bb;
        }
    }
}

// ---------------- combine: out = w0*c0 + w1*c1 ----------------
__global__ void combine_kernel(const float* __restrict__ contrib,
                               const float* __restrict__ wgt,
                               float* __restrict__ out) {
    int i  = blockIdx.x * 256 + threadIdx.x;   // over NT*(DM/4)
    int n  = i / (DM / 4);
    int c4 = (i % (DM / 4)) * 4;
    float w0 = wgt[n * 2], w1 = wgt[n * 2 + 1];
    v4f a = *(const v4f*)(contrib + ((size_t)n * 2 + 0) * DM + c4);
    v4f b = *(const v4f*)(contrib + ((size_t)n * 2 + 1) * DM + c4);
    v4f o = { w0 * a.x + w1 * b.x, w0 * a.y + w1 * b.y,
              w0 * a.z + w1 * b.z, w0 * a.w + w1 * b.w };
    *(v4f*)(out + (size_t)n * DM + c4) = o;
}

__global__ void loss_kernel(const float* __restrict__ psum,
                            const int* __restrict__ cnt, float* __restrict__ out) {
    if (threadIdx.x == 0) {
        float s = 0.f;
        for (int e = 0; e < EXP; e++)
            s += (psum[e] / (float)NT) * ((float)cnt[e] / (float)NT);
        out[(size_t)NT * DM] = s;
    }
}

// ---------------------------------------------------------------------------
extern "C" void kernel_launch(void* const* d_in, const int* in_sizes, int n_in,
                              void* d_out, int out_size, void* d_ws, size_t ws_size,
                              hipStream_t stream) {
    const float* x      = (const float*)d_in[0];
    const float* gate_w = (const float*)d_in[1];
    const float* gate_b = (const float*)d_in[2];
    const float* w1     = (const float*)d_in[3];
    const float* b1     = (const float*)d_in[4];
    const float* w2     = (const float*)d_in[5];
    const float* b2     = (const float*)d_in[6];
    float* out = (float*)d_out;

    char* ws = (char*)d_ws;
    bf16_t* x_bf     = (bf16_t*)(ws + WS_XBF);
    bf16_t* w1t      = (bf16_t*)(ws + WS_W1T);
    bf16_t* w2t      = (bf16_t*)(ws + WS_W2T);
    float*  contrib  = (float*) (ws + WS_CONTR);
    float*  wgt      = (float*) (ws + WS_WGT);
    int*    tok_list = (int*)   (ws + WS_LIST);
    int*    cnt      = (int*)   (ws + WS_CNT);
    float*  psum     = (float*) (ws + WS_PSUM);

    zero_gate_state<<<1, 32, 0, stream>>>(cnt, psum);
    cvt_x_kernel<<<(NT * DM / 4) / 256, 256, 0, stream>>>(x, x_bf);
    transpose_cvt_kernel<<<dim3(HID / 32, DM / 32, EXP), 256, 0, stream>>>(w1, w1t, DM, HID);
    transpose_cvt_kernel<<<dim3(DM / 32, HID / 32, EXP), 256, 0, stream>>>(w2, w2t, HID, DM);
    gate_kernel<<<NT / 8, 256, 0, stream>>>(x, gate_w, gate_b, tok_list, wgt, cnt, psum);
    expert_ffn_kernel<<<EXP * TILES_PER_E, 256, 0, stream>>>(
        x_bf, w1t, w2t, b1, b2, tok_list, cnt, contrib);
    combine_kernel<<<(NT * (DM / 4)) / 256, 256, 0, stream>>>(contrib, wgt, out);
    loss_kernel<<<1, 32, 0, stream>>>(psum, cnt, out);
}
